// PureE1Cell_738734375264
// MI455X (gfx1250) — matrix operand order repaired and checked by reading, not compiled, not run on hardware
//
#include <hip/hip_runtime.h>
#include <math.h>

#define T_STEPS 512
#define BATCH   32
#define DIM     1024
#define KSTEPS  (DIM / 32)          // 32 k-chunks of 32
#define RNN_BLOCKS 16               // 16 WGs x 64 cols = 1024 cols
#define WG_COLS 64                  // columns owned by one RNN workgroup
#define LDS_ELEMS (KSTEPS * WG_COLS * 32)   // 65536 bf16 = 128 KB

typedef __attribute__((ext_vector_type(16))) __bf16 v16bf;
typedef __attribute__((ext_vector_type(8)))  float  v8f;

// ---------------------------------------------------------------------------
// Grid-wide barrier via global atomics (all RNN_BLOCKS blocks are resident).
// ---------------------------------------------------------------------------
__device__ __forceinline__ void grid_barrier(int* bar, int nblk) {
    __syncthreads();
    if (threadIdx.x == 0) {
        __threadfence();
        volatile int* vgen = (volatile int*)&bar[1];
        int gen = *vgen;
        if (atomicAdd(&bar[0], 1) == nblk - 1) {
            bar[0] = 0;
            __threadfence();
            atomicAdd(&bar[1], 1);
        } else {
            while (*vgen == gen) { /* spin */ }
        }
        __threadfence();
    }
    __syncthreads();
}

// ---------------------------------------------------------------------------
// One 16x16 f32 tile: A(16 x DIM bf16 row-major) x Bpack(DIM x bcols, packed).
// 4 independent accumulators break the WMMA RAW chain; summed at the end.
//
// A-layout (contiguous k, matching Bpack): lane L holds row M=L%16;
//   lanes 0-15:  elems 0..15 -> K 0..15
//   lanes 16-31: elems 0..15 -> K 16..31
// => two b128 loads per lane per k-step.
//
// Bpack layout: pack[((kk*bcols + c)*2 + half)*16 + e] = W[c][kk*32+half*16+e]
// => lane (half = lane/16, c = bcol0 + lane%16) loads 32 contiguous bytes.
// Bp may point to global memory (GEMM) or LDS (RNN) — addrspace inferred.
// ---------------------------------------------------------------------------
__device__ __forceinline__ v8f wmma_dot(const __bf16* __restrict__ A,
                                        const __bf16* Bp, int bcols,
                                        int row0, int bcol0, int lane) {
    const int r     = row0 + (lane & 15);
    const int kbase = (lane >> 4) << 3;     // 0 or 8
    const int half  = (lane >> 4);          // 0 or 1
    const int bc    = bcol0 + (lane & 15);

    v8f acc[4];
#pragma unroll
    for (int u = 0; u < 4; ++u) acc[u] = v8f{0.f,0.f,0.f,0.f,0.f,0.f,0.f,0.f};

    for (int kk = 0; kk < KSTEPS; kk += 4) {
#pragma unroll
        for (int u = 0; u < 4; ++u) {
            union { uint4 q[2]; v16bf v; } a, b;

            const __bf16* ap = A + (size_t)r * DIM + (kk + u) * 32;
            a.q[0] = *(const uint4*)(ap + half * 16);      // K 16*half..+7
            a.q[1] = *(const uint4*)(ap + half * 16 + 8);  // K 16*half+8..+7

            const __bf16* bp =
                Bp + ((size_t)((kk + u) * bcols + bc) * 2 + half) * 16;
            b.q[0] = *(const uint4*)(bp);
            b.q[1] = *(const uint4*)(bp + 8);

            if (u == 0 && kk + 4 < KSTEPS)
                __builtin_prefetch(ap + 4 * 32, 0, 1);     // next A k-group

            acc[u] = __builtin_amdgcn_wmma_f32_16x16x32_bf16(
                         false, a.v, false, b.v, (short)0, acc[u],
                         false, false);
        }
    }
    return (acc[0] + acc[1]) + (acc[2] + acc[3]);
}

// ---------------------------------------------------------------------------
// Prep kernels
// ---------------------------------------------------------------------------
__global__ void cvt_bf16_kernel(const float* __restrict__ in,
                                __bf16* __restrict__ out, int n) {
    int i = blockIdx.x * blockDim.x + threadIdx.x;
    if (i < n) out[i] = (__bf16)in[i];
}

// pack[kk*DIM*32 + col*32 + half*16 + e] = W[col][kk*32 + half*16 + e]
__global__ void pack_w_kernel(const float* __restrict__ W,
                              __bf16* __restrict__ pack) {
    int i = blockIdx.x * blockDim.x + threadIdx.x;   // n = DIM*DIM
    int e    = i & 15;
    int half = (i >> 4) & 1;
    int col  = (i >> 5) & (DIM - 1);
    int kk   = i >> 15;
    int k    = kk * 32 + half * 16 + e;
    pack[i] = (__bf16)W[(size_t)col * DIM + k];
}

__global__ void init_kernel(const float* __restrict__ h0,
                            __bf16* __restrict__ hbuf, int* bar) {
    int i = blockIdx.x * blockDim.x + threadIdx.x;
    if (i < BATCH * DIM) hbuf[i] = (__bf16)h0[i];
    if (i < 2) bar[i] = 0;
}

// ---------------------------------------------------------------------------
// Phase 1: Wx[(t,b), e] = x[(t,b), :] . W_x[e, :]   (16384 x 1024 x 1024)
// block = 256 threads = 8 waves; each wave one 16x16 tile; B from L2.
// ---------------------------------------------------------------------------
__global__ __launch_bounds__(256) void gemm_wx_kernel(
        const __bf16* __restrict__ xb, const __bf16* __restrict__ wxpack,
        float* __restrict__ Wx) {
    const int wave = threadIdx.x >> 5, lane = threadIdx.x & 31;
    const int row0 = blockIdx.x * 32 + (wave >> 2) * 16;
    const int col0 = blockIdx.y * 64 + (wave & 3) * 16;

    v8f c = wmma_dot(xb, wxpack, DIM, row0, col0, lane);

    const int col   = col0 + (lane & 15);
    const int rbase = row0 + ((lane >> 4) << 3);
#pragma unroll
    for (int v = 0; v < 8; ++v)
        Wx[(size_t)(rbase + v) * DIM + col] = c[v];
}

// ---------------------------------------------------------------------------
// Phase 2: persistent recurrence. 16 blocks x 256 threads; block owns 64 cols.
// W_h pack chunk (128 KB) is staged into LDS once and reused for all 512
// steps (ds_load_b128 feeds the WMMA B operand). h ping-pongs in bf16;
// grid barrier per step.
// ---------------------------------------------------------------------------
__global__ __launch_bounds__(256) void rnn_kernel(
        const float* __restrict__ Wx, const float* __restrict__ z,
        const float* __restrict__ b,  const __bf16* __restrict__ whpack,
        __bf16* __restrict__ hbuf,    float* __restrict__ h_all,
        float* __restrict__ outs,     int* bar) {
    extern __shared__ __bf16 smem[];     // LDS_ELEMS bf16 = 128 KB

    const int wave = threadIdx.x >> 5, lane = threadIdx.x & 31;
    const int row0 = (wave >> 2) * 16;                       // 0 or 16
    const int nt   = (wave & 3);
    const int col0 = blockIdx.x * WG_COLS + nt * 16;         // global col base
    const int col   = col0 + (lane & 15);
    const int rbase = row0 + ((lane >> 4) << 3);
    const float bias = b[col];

    // --- stage this WG's packed W_h chunk into LDS (once) -----------------
    // global: per kk, WG_COLS*32 = 2048 contiguous elems at (kk*DIM + wg*64)*32
    {
        const size_t gbase = (size_t)blockIdx.x * WG_COLS * 32;
        for (int i = threadIdx.x; i < LDS_ELEMS / 8; i += blockDim.x) {
            int kk = i >> 8;                 // 256 uint4 per kk
            int j  = i & 255;
            const uint4* src =
                (const uint4*)(whpack + (size_t)kk * DIM * 32 + gbase) + j;
            ((uint4*)(smem + (size_t)kk * WG_COLS * 32))[j] = *src;
        }
        __syncthreads();
    }

    for (int t = 0; t < T_STEPS; ++t) {
        const int rb = t & 1, wb = rb ^ 1;
        v8f c = wmma_dot(hbuf + (size_t)rb * (BATCH * DIM),
                         smem, WG_COLS, row0, nt * 16, lane);

        const size_t tb = (size_t)t * BATCH * DIM;
#pragma unroll
        for (int v = 0; v < 8; ++v) {
            const int row   = rbase + v;
            const size_t ix = tb + (size_t)row * DIM + col;
            float pre = c[v] + Wx[ix] + bias;
            float hn  = tanhf(pre);
            hbuf[(size_t)wb * (BATCH * DIM) + (size_t)row * DIM + col] = (__bf16)hn;
            h_all[ix] = hn;
            float zv  = z[ix];
            outs[ix]  = hn * (zv / (1.f + __expf(-zv)));     // hn * silu(z)
        }
        grid_barrier(bar, gridDim.x);                        // h visible next step
    }
}

// ---------------------------------------------------------------------------
// Launch
// ---------------------------------------------------------------------------
extern "C" void kernel_launch(void* const* d_in, const int* in_sizes, int n_in,
                              void* d_out, int out_size, void* d_ws, size_t ws_size,
                              hipStream_t stream) {
    const float* x   = (const float*)d_in[0];   // [T,B,D]
    const float* z   = (const float*)d_in[1];   // [T,B,D]
    const float* h0  = (const float*)d_in[2];   // [B,D]
    const float* W_x = (const float*)d_in[3];   // [D,D]
    const float* W_h = (const float*)d_in[4];   // [D,D]
    const float* b   = (const float*)d_in[5];   // [D]

    const size_t TBD = (size_t)T_STEPS * BATCH * DIM;

    // workspace layout (256B aligned)
    char* ws = (char*)d_ws;
    size_t off = 0;
    int*    bar    = (int*)(ws + off);    off += 256;
    __bf16* hbuf   = (__bf16*)(ws + off); off += (size_t)2 * BATCH * DIM * 2;
    float*  Wx     = (float*)(ws + off);  off += TBD * 4;
    __bf16* xb     = (__bf16*)(ws + off); off += TBD * 2;
    __bf16* wxpack = (__bf16*)(ws + off); off += (size_t)DIM * DIM * 2;
    __bf16* whpack = (__bf16*)(ws + off); off += (size_t)DIM * DIM * 2;

    float* h_all = (float*)d_out;          // [T,B,D]
    float* outs  = h_all + TBD;            // [T,B,D]

    // prep
    cvt_bf16_kernel<<<(int)((TBD + 255) / 256), 256, 0, stream>>>(x, xb, (int)TBD);
    pack_w_kernel<<<(DIM * DIM) / 256, 256, 0, stream>>>(W_x, wxpack);
    pack_w_kernel<<<(DIM * DIM) / 256, 256, 0, stream>>>(W_h, whpack);
    init_kernel<<<(BATCH * DIM + 255) / 256, 256, 0, stream>>>(h0, hbuf, bar);

    // phase 1: input projection GEMM on WMMA
    dim3 g1((T_STEPS * BATCH) / 32, DIM / 64);
    gemm_wx_kernel<<<g1, 256, 0, stream>>>(xb, wxpack, Wx);

    // phase 2: sequential recurrence, persistent grid, W_h chunk in LDS
    rnn_kernel<<<RNN_BLOCKS, 256, LDS_ELEMS * sizeof(__bf16), stream>>>(
        Wx, z, b, whpack, hbuf, h_all, outs, bar);
}
